// MultiheadHollowAttention_60052232733087
// MI455X (gfx1250) — compile-verified, matrix operand order checked by reading, not run on hardware
//
#include <hip/hip_runtime.h>

// ---------------------------------------------------------------------------
// MultiheadHollowAttention on MI455X (gfx1250).
//
// The hollow mask keeps ONLY the diagonal -> softmax == identity -> the whole
// attention is   out = (V @ Wv^T + bv) @ Wo^T + bo.
// Fold weights:  Wf = Wo @ Wv  (so out = V @ Wf^T + (Wo@bv + bo)).
// Two WMMA GEMMs total (~10.7 GFLOP), ~44 MB HBM traffic => bandwidth bound.
// ---------------------------------------------------------------------------

typedef __attribute__((ext_vector_type(16))) _Float16 v16h;
typedef __attribute__((ext_vector_type(8)))  _Float16 v8h;
typedef __attribute__((ext_vector_type(4)))  _Float16 v4h;
typedef __attribute__((ext_vector_type(8)))  float    v8f;

#define D_DIM   1024          // model dim (K and N of both GEMMs)
#define M_ROWS  4096          // B*S tokens
#define TS      128           // block tile (M and N)
#define KBLK    32            // K step == WMMA K
#define LDS_PAD 8             // halfs; keeps rows 16B-aligned (stride 80B)
#define LSTRIDE (KBLK + LDS_PAD)   // 40 halfs per LDS tile row

// Per-lane WMMA A/B fragment gather from an LDS tile (row-major, K-contig).
// ISA 7.12.2 (16-bit 16x32 A layout): lane = 16*hi + m holds row m,
// halfs [0..7] = K[8*hi .. 8*hi+7], halfs [8..15] = K[16+8*hi .. 16+8*hi+7].
__device__ __forceinline__ v16h load_frag(const _Float16* __restrict__ tile,
                                          int lane) {
    const int m  = lane & 15;
    const int hi = lane >> 4;
    const _Float16* row = tile + m * LSTRIDE;
    v16h f;
    *((v8h*)&f)     = *(const v8h*)(row + 8 * hi);        // ds_load_b128
    *((v8h*)&f + 1) = *(const v8h*)(row + 16 + 8 * hi);   // ds_load_b128
    return f;
}

__device__ __forceinline__ v8f wmma_f16(v16h a, v16h b, v8f c) {
    return __builtin_amdgcn_wmma_f32_16x16x32_f16(
        /*neg_a=*/false, a, /*neg_b=*/false, b,
        /*c_mod=*/(short)0, c, /*reuse_a=*/false, /*reuse_b=*/false);
}

// ---------------------------------------------------------------------------
// Kernel 0: Wf[i][j] = sum_k Wo[i][k] * Wv[k][j]   (1024x1024x1024, f16 out)
// A = Wo row-major (K-contig, direct). B operand needs column j of Wv
// K-contiguous -> stage Wv tile transposed into LDS.
// ---------------------------------------------------------------------------
__global__ __launch_bounds__(256)
void fuse_weights_kernel(const float* __restrict__ Wo,
                         const float* __restrict__ Wv,
                         _Float16* __restrict__ Wf) {
    __shared__ __align__(16) _Float16 ldsA[TS * LSTRIDE];
    __shared__ __align__(16) _Float16 ldsB[TS * LSTRIDE];

    const int t     = threadIdx.x;
    const int lane  = t & 31;
    const int wave  = t >> 5;       // 0..7
    const int waveM = wave >> 2;    // 0..1  (64 rows of i)
    const int waveN = wave & 3;     // 0..3  (32 cols of j)
    const int bI    = blockIdx.x * TS;
    const int bJ    = blockIdx.y * TS;

    v8f acc[4][2];
#pragma unroll
    for (int i = 0; i < 4; ++i)
#pragma unroll
        for (int j = 0; j < 2; ++j)
#pragma unroll
            for (int e = 0; e < 8; ++e) acc[i][j][e] = 0.0f;

    for (int kb = 0; kb < D_DIM; kb += KBLK) {
        __syncthreads();
        // A tile: 128 rows(i) x 32 k, f32 -> f16. 1024 float4 chunks / 256 thr.
#pragma unroll
        for (int it = 0; it < 4; ++it) {
            const int idx = t + it * 256;
            const int row = idx >> 3;      // /8 float4 per row
            const int c4  = idx & 7;
            const float4 f = *(const float4*)(Wo + (size_t)(bI + row) * D_DIM + kb + c4 * 4);
            v4h h = { (_Float16)f.x, (_Float16)f.y, (_Float16)f.z, (_Float16)f.w };
            *(v4h*)(ldsA + row * LSTRIDE + c4 * 4) = h;
        }
        // B tile (transposed): ldsB[j][k] = Wv[kb+k][bJ+j]. Coalesced reads
        // along j, scattered 2B LDS stores. 32 k-rows x 32 float4 of j.
#pragma unroll
        for (int it = 0; it < 4; ++it) {
            const int idx = t + it * 256;
            const int k  = idx >> 5;       // /32
            const int c4 = idx & 31;
            const float4 f = *(const float4*)(Wv + (size_t)(kb + k) * D_DIM + bJ + c4 * 4);
            const int j0 = c4 * 4;
            ldsB[(j0 + 0) * LSTRIDE + k] = (_Float16)f.x;
            ldsB[(j0 + 1) * LSTRIDE + k] = (_Float16)f.y;
            ldsB[(j0 + 2) * LSTRIDE + k] = (_Float16)f.z;
            ldsB[(j0 + 3) * LSTRIDE + k] = (_Float16)f.w;
        }
        __syncthreads();

        v16h af[4], bf[2];
#pragma unroll
        for (int i = 0; i < 4; ++i)
            af[i] = load_frag(ldsA + (waveM * 64 + i * 16) * LSTRIDE, lane);
#pragma unroll
        for (int j = 0; j < 2; ++j)
            bf[j] = load_frag(ldsB + (waveN * 32 + j * 16) * LSTRIDE, lane);
#pragma unroll
        for (int i = 0; i < 4; ++i)
#pragma unroll
            for (int j = 0; j < 2; ++j)
                acc[i][j] = wmma_f16(af[i], bf[j], acc[i][j]);
    }

    // Epilogue: C/D layout = lane covers one column (j), 8 rows (i) per vgpr.
    const int n  = lane & 15;
    const int hi = lane >> 4;
#pragma unroll
    for (int j = 0; j < 2; ++j) {
        const int col = bJ + waveN * 32 + j * 16 + n;
#pragma unroll
        for (int i = 0; i < 4; ++i) {
            const int rowBase = bI + waveM * 64 + i * 16 + hi * 8;
#pragma unroll
            for (int p = 0; p < 8; ++p)
                Wf[(size_t)(rowBase + p) * D_DIM + col] = (_Float16)acc[i][j][p];
        }
    }
}

// ---------------------------------------------------------------------------
// Fused bias: bf[i] = dot(Wo[i,:], bv) + bo[i]   (trivial, 1M MACs)
// ---------------------------------------------------------------------------
__global__ __launch_bounds__(256)
void fuse_bias_kernel(const float* __restrict__ Wo,
                      const float* __restrict__ bv,
                      const float* __restrict__ bo,
                      float* __restrict__ bf) {
    const int i = blockIdx.x * blockDim.x + threadIdx.x;
    if (i < D_DIM) {
        float s = bo[i];
        const float* row = Wo + (size_t)i * D_DIM;
        for (int k = 0; k < D_DIM; ++k) s = fmaf(row[k], bv[k], s);
        bf[i] = s;
    }
}

// ---------------------------------------------------------------------------
// Kernel 1: out[n][i] = sum_j V[n][j] * Wf[i][j] + bf[i]
// A = V (f32 -> f16 on the fly). B operand = Wf rows, already K-contig f16.
// ---------------------------------------------------------------------------
__global__ __launch_bounds__(256)
void gemm_vwf_kernel(const float* __restrict__ V,
                     const _Float16* __restrict__ Wf,
                     const float* __restrict__ bias,
                     float* __restrict__ out) {
    __shared__ __align__(16) _Float16 ldsA[TS * LSTRIDE];
    __shared__ __align__(16) _Float16 ldsB[TS * LSTRIDE];

    const int t     = threadIdx.x;
    const int lane  = t & 31;
    const int wave  = t >> 5;
    const int waveM = wave >> 2;    // 0..1
    const int waveN = wave & 3;     // 0..3
    const int bM    = blockIdx.x * TS;   // token rows
    const int bN    = blockIdx.y * TS;   // output features

    v8f acc[4][2];
#pragma unroll
    for (int i = 0; i < 4; ++i)
#pragma unroll
        for (int j = 0; j < 2; ++j)
#pragma unroll
            for (int e = 0; e < 8; ++e) acc[i][j][e] = 0.0f;

    for (int kb = 0; kb < D_DIM; kb += KBLK) {
        __syncthreads();
        // A tile: 128x32 from V (f32 -> f16)
#pragma unroll
        for (int it = 0; it < 4; ++it) {
            const int idx = t + it * 256;
            const int row = idx >> 3;
            const int c4  = idx & 7;
            const float4 f = *(const float4*)(V + (size_t)(bM + row) * D_DIM + kb + c4 * 4);
            v4h h = { (_Float16)f.x, (_Float16)f.y, (_Float16)f.z, (_Float16)f.w };
            *(v4h*)(ldsA + row * LSTRIDE + c4 * 4) = h;
        }
        // B tile: 128x32 halfs straight from Wf (row-major, K-contig)
#pragma unroll
        for (int it = 0; it < 4; ++it) {
            const int idx = t + it * 256;
            const int row = idx >> 3;
            const int c   = idx & 7;
            v4h h = *(const v4h*)(Wf + (size_t)(bN + row) * D_DIM + kb + c * 4);
            *(v4h*)(ldsB + row * LSTRIDE + c * 4) = h;
        }
        __syncthreads();

        v16h af[4], bfg[2];
#pragma unroll
        for (int i = 0; i < 4; ++i)
            af[i] = load_frag(ldsA + (waveM * 64 + i * 16) * LSTRIDE, lane);
#pragma unroll
        for (int j = 0; j < 2; ++j)
            bfg[j] = load_frag(ldsB + (waveN * 32 + j * 16) * LSTRIDE, lane);
#pragma unroll
        for (int i = 0; i < 4; ++i)
#pragma unroll
            for (int j = 0; j < 2; ++j)
                acc[i][j] = wmma_f16(af[i], bfg[j], acc[i][j]);
    }

    // Epilogue: per-lane column is fixed -> one bias load per 16x16 tile.
    const int n  = lane & 15;
    const int hi = lane >> 4;
#pragma unroll
    for (int j = 0; j < 2; ++j) {
        const int col = bN + waveN * 32 + j * 16 + n;
        const float bv = bias[col];
#pragma unroll
        for (int i = 0; i < 4; ++i) {
            const int rowBase = bM + waveM * 64 + i * 16 + hi * 8;
#pragma unroll
            for (int p = 0; p < 8; ++p)
                out[(size_t)(rowBase + p) * D_DIM + col] = acc[i][j][p] + bv;
        }
    }
}

// ---------------------------------------------------------------------------
// Inputs (setup_inputs order):
//  0 query  1 key  2 value  3 mask  4 Wq  5 bq  6 Wk  7 bk  8 Wv  9 bv
// 10 Wo    11 bo
// Only value, Wv, bv, Wo, bo are live (diagonal-only attention).
// ---------------------------------------------------------------------------
extern "C" void kernel_launch(void* const* d_in, const int* in_sizes, int n_in,
                              void* d_out, int out_size, void* d_ws, size_t ws_size,
                              hipStream_t stream) {
    const float* V  = (const float*)d_in[2];
    const float* Wv = (const float*)d_in[8];
    const float* bv = (const float*)d_in[9];
    const float* Wo = (const float*)d_in[10];
    const float* bo = (const float*)d_in[11];
    float* out = (float*)d_out;

    // Workspace layout: [0, 2MB) Wf (f16 1024x1024), then fused bias (f32 1024)
    _Float16* Wf = (_Float16*)d_ws;
    float*    bf = (float*)((char*)d_ws + (size_t)D_DIM * D_DIM * sizeof(_Float16));

    fuse_weights_kernel<<<dim3(D_DIM / TS, D_DIM / TS), 256, 0, stream>>>(Wo, Wv, Wf);
    fuse_bias_kernel<<<(D_DIM + 255) / 256, 256, 0, stream>>>(Wo, bv, bo, bf);
    gemm_vwf_kernel<<<dim3(M_ROWS / TS, D_DIM / TS), 256, 0, stream>>>(V, Wf, bf, out);
}
